// GAT_71889162600962
// MI455X (gfx1250) — matrix operand order repaired and checked by reading, not compiled, not run on hardware
//
#include <hip/hip_runtime.h>
#include <hip/hip_bf16.h>

// ---------------------------------------------------------------------------
// 2-layer sparse GAT for MI455X (gfx1250, wave32).
//  - GEMM h@W via v_wmma_f32_16x16x32_bf16 (bf16 inputs, f32 accumulate)
//  - edge aggregation via global_atomic_add_f32 (L2-resident: h2/num ~102MB
//    each fit in the 192MB L2, so the 3.4GB/layer gather stays on-chip)
// ---------------------------------------------------------------------------

#define NN 100000
#define DD 256
#define ND (NN * DD)
#define NEG_SLOPE 0.2f

typedef __attribute__((ext_vector_type(16))) __bf16 v16bf;
typedef __attribute__((ext_vector_type(8)))  __bf16 v8bf;
typedef __attribute__((ext_vector_type(8)))  float  v8f;

__device__ __forceinline__ unsigned short f32_to_bf16_rne(float f) {
    unsigned int u = __float_as_uint(f);
    unsigned int rounding = 0x7FFFu + ((u >> 16) & 1u);
    return (unsigned short)((u + rounding) >> 16);
}

// ---- cast h (row-major) to bf16 -------------------------------------------
__global__ void gat_cast_h_bf16(const float* __restrict__ in,
                                unsigned short* __restrict__ out, int n) {
    int i = blockIdx.x * blockDim.x + threadIdx.x;
    if (i < n) out[i] = f32_to_bf16_rne(in[i]);
}

// ---- cast W to bf16, transposed (column-major) so B fragments are contiguous
__global__ void gat_cast_w_bf16_t(const float* __restrict__ W,
                                  unsigned short* __restrict__ WT) {
    int i = blockIdx.x * blockDim.x + threadIdx.x;   // over DD*DD
    int k = i >> 8;          // row of W
    int n = i & 255;         // col of W
    WT[(size_t)n * DD + k] = f32_to_bf16_rne(W[i]);
}

// ---- GEMM: C[N,D] = A[N,D] (bf16) @ B[D,D] (bf16, given transposed) --------
// One wave computes one 16x16 C tile; K-loop in steps of 32 via WMMA.
__global__ void gat_gemm_wmma_bf16(const unsigned short* __restrict__ Ab,
                                   const unsigned short* __restrict__ BTb,
                                   float* __restrict__ C, int nrows) {
    const int wave = (blockIdx.x * blockDim.x + threadIdx.x) >> 5;
    const int lane = threadIdx.x & 31;
    const int ntc  = DD / 16;                 // 16 column tiles
    const int tileRow = wave / ntc;
    const int tileCol = wave % ntc;
    if (tileRow * 16 >= nrows) return;

    const int half = lane >> 4;               // 0: lanes 0-15, 1: lanes 16-31
    const int l15  = lane & 15;

    // A fragment (16x32 bf16): lane half=0 holds K 0..7 & 16..23 of row l15;
    // half=1 holds K 8..15 & 24..31.  Two 16B loads per lane.
    const __bf16* arow = (const __bf16*)(Ab + (size_t)(tileRow * 16 + l15) * DD);
    // B fragment (32x16 bf16): lane = column n (l15); half=0 holds K 0..7 &
    // 16..23, half=1 holds K 8..15 & 24..31 (matching A). BT is column-major W.
    const __bf16* bcol = (const __bf16*)(BTb + (size_t)(tileCol * 16 + l15) * DD);

    v8f acc = {};
    for (int k = 0; k < DD; k += 32) {
        if (k + 32 < DD) {  // prefetch next K slice -> global_prefetch_b8
            __builtin_prefetch((const void*)(arow + k + 32 + half * 8), 0, 3);
            __builtin_prefetch((const void*)(bcol + k + 32 + half * 16), 0, 3);
        }
        v8bf a_lo = *(const v8bf*)(arow + k + half * 8);
        v8bf a_hi = *(const v8bf*)(arow + k + 16 + half * 8);
        v8bf b_lo = *(const v8bf*)(bcol + k + half * 8);
        v8bf b_hi = *(const v8bf*)(bcol + k + 16 + half * 8);
        v16bf av = __builtin_shufflevector(a_lo, a_hi,
                     0,1,2,3,4,5,6,7,8,9,10,11,12,13,14,15);
        v16bf bv = __builtin_shufflevector(b_lo, b_hi,
                     0,1,2,3,4,5,6,7,8,9,10,11,12,13,14,15);
        acc = __builtin_amdgcn_wmma_f32_16x16x32_bf16(
                  false, av, false, bv, (short)0, acc, false, false);
    }

    // C fragment layout: lane -> column n=l15; VGPR v -> row half*8+v.
    const int n = l15;
    const int mbase = half * 8;
    #pragma unroll
    for (int v = 0; v < 8; ++v) {
        C[(size_t)(tileRow * 16 + mbase + v) * DD + tileCol * 16 + n] = acc[v];
    }
}

// ---- per-row scores: s_src = h2 . a[:D], s_dst = h2 . a[D:] ---------------
__global__ void gat_row_scores(const float* __restrict__ h2,
                               const float* __restrict__ a,
                               float* __restrict__ ssrc,
                               float* __restrict__ sdst, int nrows) {
    const int row  = (blockIdx.x * blockDim.x + threadIdx.x) >> 5;
    const int lane = threadIdx.x & 31;
    if (row >= nrows) return;
    const float* r = h2 + (size_t)row * DD;
    float p0 = 0.f, p1 = 0.f;
    #pragma unroll
    for (int j = 0; j < 8; ++j) {
        int c = lane + j * 32;                 // coalesced
        float v = r[c];
        p0 += v * a[c];
        p1 += v * a[DD + c];
    }
    #pragma unroll
    for (int off = 16; off > 0; off >>= 1) {
        p0 += __shfl_down(p0, off, 32);
        p1 += __shfl_down(p1, off, 32);
    }
    if (lane == 0) { ssrc[row] = p0; sdst[row] = p1; }
}

// ---- zero accumulators -----------------------------------------------------
__global__ void gat_zero(float* __restrict__ num, float* __restrict__ denom,
                         int nnum, int ndenom) {
    int i = blockIdx.x * blockDim.x + threadIdx.x;
    if (i < nnum) num[i] = 0.f;
    if (i < ndenom) denom[i] = 0.f;
}

// ---- edge aggregation: one wave per edge, 8 dims per lane ------------------
__global__ void gat_edge_agg(const int* __restrict__ src,
                             const int* __restrict__ dst,
                             const float* __restrict__ ssrc,
                             const float* __restrict__ sdst,
                             const float* __restrict__ h2,
                             float* __restrict__ denom,
                             float* __restrict__ num, int E) {
    const int e    = (blockIdx.x * blockDim.x + threadIdx.x) >> 5;
    const int lane = threadIdx.x & 31;
    if (e >= E) return;
    const int s = src[e];
    const int d = dst[e];
    float score = ssrc[s] + sdst[d];
    float lr = score > 0.f ? score : NEG_SLOPE * score;
    float w = __expf(-lr);
    if (lane == 0) atomicAdd(denom + s, w);
    const float* hd = h2 + (size_t)d * DD;
    float* ns = num + (size_t)s * DD;
    #pragma unroll
    for (int j = 0; j < 8; ++j) {
        int c = lane + j * 32;                 // coalesced gather + atomic
        atomicAdd(ns + c, w * hd[c]);
    }
}

// ---- finalize: out = elu(num / denom[row]) in place ------------------------
__global__ void gat_finalize_elu(float* __restrict__ num,
                                 const float* __restrict__ denom, int n) {
    int i = blockIdx.x * blockDim.x + threadIdx.x;
    if (i >= n) return;
    int row = i >> 8;                          // DD == 256
    float x = num[i] / denom[row];
    num[i] = x > 0.f ? x : (__expf(x) - 1.f);
}

// ---------------------------------------------------------------------------
static void run_layer(const float* h_in, const float* W, const float* a,
                      const int* src, const int* dst, int E,
                      float* h2, float* out /* num, finalized in place */,
                      unsigned short* Hb, unsigned short* WTb,
                      float* ssrc, float* sdst, float* denom,
                      hipStream_t stream) {
    const int B = 256;
    gat_cast_h_bf16 <<<ND / B, B, 0, stream>>>(h_in, Hb, ND);
    gat_cast_w_bf16_t<<<(DD * DD) / B, B, 0, stream>>>(W, WTb);
    gat_gemm_wmma_bf16<<<(NN / 16) * (DD / 16) / 8, B, 0, stream>>>(Hb, WTb, h2, NN);
    gat_row_scores  <<<NN / 8, B, 0, stream>>>(h2, a, ssrc, sdst, NN);
    gat_zero        <<<ND / B, B, 0, stream>>>(out, denom, ND, NN);
    gat_edge_agg    <<<(E + 7) / 8, B, 0, stream>>>(src, dst, ssrc, sdst,
                                                    h2, denom, out, E);
    gat_finalize_elu<<<ND / B, B, 0, stream>>>(out, denom, ND);
}

extern "C" void kernel_launch(void* const* d_in, const int* in_sizes, int n_in,
                              void* d_out, int out_size, void* d_ws, size_t ws_size,
                              hipStream_t stream) {
    const float* emb = (const float*)d_in[0];
    const float* W1  = (const float*)d_in[1];
    const float* a1  = (const float*)d_in[2];
    const float* W2  = (const float*)d_in[3];
    const float* a2  = (const float*)d_in[4];
    const int* edges = (const int*)d_in[5];
    const int E      = in_sizes[5] / 2;
    const int* src   = edges;
    const int* dst   = edges + E;

    // Workspace layout (~257 MB):
    float* h2    = (float*)d_ws;                       // ND f32
    float* hmid  = h2 + (size_t)ND;                    // ND f32 (layer1 num/out)
    float* ssrc  = hmid + (size_t)ND;                  // NN
    float* sdst  = ssrc + NN;                          // NN
    float* denom = sdst + NN;                          // NN
    unsigned short* Hb  = (unsigned short*)(denom + NN);   // ND bf16
    unsigned short* WTb = Hb + (size_t)ND;                 // DD*DD bf16

    // Layer 1: emb -> hmid
    run_layer(emb, W1, a1, src, dst, E, h2, hmid, Hb, WTb, ssrc, sdst, denom, stream);
    // Layer 2: hmid -> d_out
    run_layer(hmid, W2, a2, src, dst, E, h2, (float*)d_out, Hb, WTb, ssrc, sdst, denom, stream);
}
